// CdfgReader_28321014350505
// MI455X (gfx1250) — compile-verified, matrix operand order checked by reading, not run on hardware
//
#include <hip/hip_runtime.h>
#include <hip/hip_bf16.h>

// ---------------------------------------------------------------------------
// CDFG GCN stack on MI455X (gfx1250, wave32, WMMA), LDS-free GEMM design.
//
// - Compute the stack once per CDFG (32) instead of per batch item (64).
// - One-shot pre-passes convert A/X to row-major bf16 and pack W into the
//   WMMA B-fragment ("block-packed") layout.
// - GEMMs load A and B fragments *directly* from global memory:
//     A-frag: two global b128 per lane (row-major bf16).
//     B-frag: two coalesced global b128 per lane (block-packed: the layout a
//             WMMA D fragment naturally has, so the X@W GEMM emits T already
//             fragment-ready for the A@T GEMM).
// - No LDS, no barriers, no per-K-step f32->bf16 conversion.
// ---------------------------------------------------------------------------

#define NC   32    // number of CDFGs
#define NN   1024  // nodes per CDFG
#define FEAT 128
#define HID  256
#define NL   4
#define NB   64    // batch

typedef __attribute__((ext_vector_type(16))) __bf16       v16bf;
typedef __attribute__((ext_vector_type(8)))  float        v8f;
typedef __attribute__((ext_vector_type(4)))  unsigned int u32x4;
typedef __attribute__((ext_vector_type(2)))  unsigned int u32x2;
typedef __attribute__((ext_vector_type(4)))  float        f32x4;

union FragU {
    v16bf v;
    u32x4 u[2];
};

__device__ __forceinline__ unsigned short f32_to_bf16_rne(float f) {
    unsigned int u = __float_as_uint(f);
    u += 0x7FFFu + ((u >> 16) & 1u);   // round to nearest even
    return (unsigned short)(u >> 16);
}

__device__ __forceinline__ unsigned int pack2_bf16(float a, float b) {
    unsigned int ua = __float_as_uint(a);
    ua += 0x7FFFu + ((ua >> 16) & 1u);
    unsigned int ub = __float_as_uint(b);
    ub += 0x7FFFu + ((ub >> 16) & 1u);
    return (ua >> 16) | (ub & 0xFFFF0000u);
}

__device__ __forceinline__ float bf16_to_f32(unsigned short u) {
    return __uint_as_float((unsigned int)u << 16);
}

// --------------------------- pre-pass kernels ------------------------------

// Row-major f32 -> bf16, 4 elements per thread.
__global__ __launch_bounds__(256)
void cvt_f32_to_bf16_rm(const float* __restrict__ in,
                        unsigned short* __restrict__ out, long long n4) {
    long long i = (long long)blockIdx.x * 256 + threadIdx.x;
    if (i >= n4) return;
    f32x4 x = *(const f32x4*)(in + i * 4);
    u32x2 p = { pack2_bf16(x.x, x.y), pack2_bf16(x.z, x.w) };
    *(u32x2*)(out + i * 4) = p;
}

// Pack a [K][N] f32 matrix into WMMA-B-fragment blocks of 16x16 bf16:
// element (k,n) -> block (k/16, n/16), lane = (n%16) + 16*((k%16)/8), element r = k%8.
__global__ __launch_bounds__(256)
void pack_b_f32_to_bf16(const float* __restrict__ W,
                        unsigned short* __restrict__ out, int K, int N) {
    const int blk = blockIdx.x * 8 + (threadIdx.x >> 5);
    const int NBk = N >> 4, KBk = K >> 4;
    if (blk >= KBk * NBk) return;
    const int kb = blk / NBk, nb = blk % NBk;
    const int lane = threadIdx.x & 31, idx = lane & 15, half = lane >> 4;
    float v[8];
    #pragma unroll
    for (int r = 0; r < 8; ++r)
        v[r] = W[(kb * 16 + half * 8 + r) * N + nb * 16 + idx];
    u32x4 p = { pack2_bf16(v[0], v[1]), pack2_bf16(v[2], v[3]),
                pack2_bf16(v[4], v[5]), pack2_bf16(v[6], v[7]) };
    *(u32x4*)(out + ((long long)blk * 32 + lane) * 8) = p;
}

// ------------------------------- GEMM --------------------------------------
// MODE 0: v = relu(acc+bias); write H0 (f32 block-packed) and Hc (bf16 RM)
// MODE 1: v = acc;            write T  (bf16 block-packed)
// MODE 2: v = relu(acc+bias); write Hc (bf16 RM)
// MODE 3: v = tanh(acc+bias) + H0(block-packed); write Hc (bf16 RM)
template <int MODE>
__global__ __launch_bounds__(256)
void gemm_wmma_direct(const unsigned short* __restrict__ Abf, long long aGS, int lda,
                      const unsigned short* __restrict__ Bpk, long long bGS,
                      int K,
                      const float* __restrict__ bias,
                      float* __restrict__ h0pk,           // f32 block-packed
                      unsigned short* __restrict__ outRM, // bf16 row-major
                      unsigned short* __restrict__ outPk) // bf16 block-packed
{
    const int tid  = threadIdx.x;
    const int lane = tid & 31;
    const int wave = tid >> 5;      // 0..7 -> 16-row M strip
    const int idx  = lane & 15;
    const int half = lane >> 4;

    const int m0 = blockIdx.x * 128;
    const int n0 = blockIdx.y * 128;
    const int g  = blockIdx.z;
    const int R  = m0 + wave * 16;
    const int nb0 = n0 >> 4;

    const unsigned short* __restrict__ Ag = Abf + (long long)g * aGS;
    const unsigned short* __restrict__ Bg = Bpk + (long long)g * bGS;

    // Per-lane A row pointer: row = R + idx, K sub-chunk selected by half.
    const unsigned short* __restrict__ arow =
        Ag + (long long)(R + idx) * lda + 8 * half;

    v8f acc[8] = {};

    for (int kk = 0; kk < K; kk += 32) {
        FragU af;
        af.u[0] = *(const u32x4*)(arow + kk);        // K = kk + 8*half .. +7
        af.u[1] = *(const u32x4*)(arow + kk + 16);   // K = kk + 16 + 8*half .. +7
        if (kk + 32 < K)
            __builtin_prefetch(arow + kk + 32, 0, 3);

        // Block-packed B: block (kb, nb) at ((kb*16 + nb) * 512B), lane-contiguous.
        const unsigned short* __restrict__ bbase =
            Bg + (((long long)(kk >> 4) * 16 + nb0) * 32 + lane) * 8;
        #pragma unroll
        for (int j = 0; j < 8; ++j) {
            FragU bf;
            bf.u[0] = *(const u32x4*)(bbase + j * 256);              // k-block kb
            bf.u[1] = *(const u32x4*)(bbase + 16 * 256 + j * 256);   // k-block kb+1
            acc[j] = __builtin_amdgcn_wmma_f32_16x16x32_bf16(
                false, af.v, false, bf.v, (short)0, acc[j], false, false);
        }
    }

    // ---- epilogue ----
    const long long gElems = (long long)g * NN * HID;  // per-graph elements (all layouts)
    const int rowblk = R >> 4;

    #pragma unroll
    for (int j = 0; j < 8; ++j) {
        const int gcol   = n0 + j * 16 + idx;
        const int colblk = nb0 + j;
        const long long pkOff = gElems + ((long long)(rowblk * 16 + colblk) * 32 + lane) * 8;

        float bv = 0.0f;
        if (MODE != 1) bv = bias[gcol];

        if (MODE == 1) {
            u32x4 p = { pack2_bf16(acc[j][0], acc[j][1]), pack2_bf16(acc[j][2], acc[j][3]),
                        pack2_bf16(acc[j][4], acc[j][5]), pack2_bf16(acc[j][6], acc[j][7]) };
            *(u32x4*)(outPk + pkOff) = p;
        } else {
            float v[8];
            if (MODE == 3) {
                f32x4 r0 = *(const f32x4*)(h0pk + pkOff);
                f32x4 r1 = *(const f32x4*)(h0pk + pkOff + 4);
                #pragma unroll
                for (int r = 0; r < 8; ++r) {
                    float res = (r < 4) ? r0[r] : r1[r - 4];
                    v[r] = tanhf(acc[j][r] + bv) + res;
                }
            } else {
                #pragma unroll
                for (int r = 0; r < 8; ++r)
                    v[r] = fmaxf(acc[j][r] + bv, 0.0f);
            }
            if (MODE == 0) {
                f32x4 a = { v[0], v[1], v[2], v[3] };
                f32x4 b = { v[4], v[5], v[6], v[7] };
                *(f32x4*)(h0pk + pkOff)     = a;
                *(f32x4*)(h0pk + pkOff + 4) = b;
            }
            #pragma unroll
            for (int r = 0; r < 8; ++r) {
                const int grow = R + 8 * half + r;
                outRM[gElems + (long long)grow * HID + gcol] = f32_to_bf16_rne(v[r]);
            }
        }
    }
}

// Final masked mean from bf16 row-major Hc.
__global__ __launch_bounds__(256)
void masked_mean_kernel(const unsigned short* __restrict__ Hc,
                        const int* __restrict__ graph,
                        const unsigned char* __restrict__ mask,
                        float* __restrict__ out) {
    const int b = blockIdx.x;
    const int h = threadIdx.x;  // 0..255
    const int g = graph[b];
    const unsigned short* __restrict__ Hg = Hc + (long long)g * NN * HID;
    const unsigned char*  __restrict__ mb = mask + (long long)b * NN;
    float s = 0.0f, cnt = 0.0f;
    for (int n = 0; n < NN; ++n) {
        if (mb[n]) {
            s += bf16_to_f32(Hg[n * HID + h]);
            cnt += 1.0f;
        }
    }
    out[b * HID + h] = s / cnt;
}

// ------------------------------ launcher -----------------------------------

extern "C" void kernel_launch(void* const* d_in, const int* in_sizes, int n_in,
                              void* d_out, int out_size, void* d_ws, size_t ws_size,
                              hipStream_t stream) {
    const float*         cdfg_xs = (const float*)d_in[0];        // [32,1024,128]
    const float*         cdfg_as = (const float*)d_in[1];        // [32,1024,1024]
    const int*           graph   = (const int*)d_in[2];          // [64]
    const unsigned char* cmask   = (const unsigned char*)d_in[3];// [64,1024] bool
    const float*         W_in    = (const float*)d_in[4];        // [128,256]
    const float*         b_in    = (const float*)d_in[5];        // [256]
    const float*         W_gcn   = (const float*)d_in[6];        // [4,256,256]
    const float*         b_gcn   = (const float*)d_in[7];        // [4,256]
    float*               out     = (float*)d_out;                // [64,256]

    // Workspace carve-up (~137 MB total).
    char* w = (char*)d_ws;
    unsigned short* Ab   = (unsigned short*)w;  w += (long long)NC * NN * NN  * 2; // 64 MB bf16 RM
    unsigned short* Xb   = (unsigned short*)w;  w += (long long)NC * NN * FEAT * 2; //  8 MB bf16 RM
    unsigned short* Winb = (unsigned short*)w;  w += (long long)FEAT * HID * 2;     // 64 KB packed
    unsigned short* Wgb  = (unsigned short*)w;  w += (long long)NL * HID * HID * 2; // 512 KB packed
    float*          H0pk = (float*)w;           w += (long long)NC * NN * HID * 4;  // 32 MB f32 packed
    unsigned short* Hc   = (unsigned short*)w;  w += (long long)NC * NN * HID * 2;  // 16 MB bf16 RM
    unsigned short* Tpk  = (unsigned short*)w;  w += (long long)NC * NN * HID * 2;  // 16 MB bf16 packed

    // ---- pre-passes: one-shot conversions ----
    {
        long long n4 = (long long)NC * NN * NN / 4;
        cvt_f32_to_bf16_rm<<<(unsigned)((n4 + 255) / 256), 256, 0, stream>>>(cdfg_as, Ab, n4);
    }
    {
        long long n4 = (long long)NC * NN * FEAT / 4;
        cvt_f32_to_bf16_rm<<<(unsigned)((n4 + 255) / 256), 256, 0, stream>>>(cdfg_xs, Xb, n4);
    }
    pack_b_f32_to_bf16<<<(FEAT / 16) * (HID / 16) / 8, 256, 0, stream>>>(W_in, Winb, FEAT, HID);
    for (int i = 0; i < NL; ++i)
        pack_b_f32_to_bf16<<<(HID / 16) * (HID / 16) / 8, 256, 0, stream>>>(
            W_gcn + (long long)i * HID * HID, Wgb + (long long)i * HID * HID, HID, HID);

    dim3 grid(NN / 128, HID / 128, NC);  // 8 x 2 x 32
    dim3 blk(256);

    // Input layer: H0, Hc = relu(X @ W_in + b_in)         (K = 128)
    gemm_wmma_direct<0><<<grid, blk, 0, stream>>>(
        Xb, (long long)NN * FEAT, FEAT,
        Winb, 0LL, FEAT, b_in, H0pk, Hc, nullptr);

    for (int i = 0; i < NL; ++i) {
        // T = Hc @ W_gcn[i]                               (K = 256)
        gemm_wmma_direct<1><<<grid, blk, 0, stream>>>(
            Hc, (long long)NN * HID, HID,
            Wgb + (long long)i * HID * HID, 0LL, HID,
            nullptr, nullptr, nullptr, Tpk);

        if (i < NL - 1) {
            // Hc = relu(A @ T + b_gcn[i])                 (K = 1024)
            gemm_wmma_direct<2><<<grid, blk, 0, stream>>>(
                Ab, (long long)NN * NN, NN,
                Tpk, (long long)NN * HID, NN,
                b_gcn + (long long)i * HID, nullptr, Hc, nullptr);
        } else {
            // Hc = tanh(A @ T + b_gcn[i]) + H0            (K = 1024)
            gemm_wmma_direct<3><<<grid, blk, 0, stream>>>(
                Ab, (long long)NN * NN, NN,
                Tpk, (long long)NN * HID, NN,
                b_gcn + (long long)i * HID, H0pk, Hc, nullptr);
        }
    }

    masked_mean_kernel<<<NB, 256, 0, stream>>>(Hc, graph, cmask, out);
}
